// CrossAttention_81235011437201
// MI455X (gfx1250) — matrix operand order repaired and checked by reading, not compiled, not run on hardware
//
#include <hip/hip_runtime.h>
#include <hip/hip_bf16.h>

// Problem sizes (fixed by the reference)
#define BATCHES 8192
#define NNODES  524288
#define HDIM    256
#define KDIM    128
#define VDIM    128
#define QDIM    128
#define SCALE   0.08838834764831845f   // 1/sqrt(128)

typedef __attribute__((ext_vector_type(16))) __bf16 v16bf;
typedef __attribute__((ext_vector_type(8)))  float  v8f;

__device__ __forceinline__ int lane_id() { return threadIdx.x & 31; }

// ---------------- WMMA fragment loaders (ISA 7.12.2, wave32) ----------------
// A: 16x32 bf16. lane l: row r = l&15, half h = l>>4.
// VGPR j (j<4): K = h*8 + 2j, +1 ; VGPR j (j>=4): K = 16 + h*8 + 2(j-4), +1.
__device__ __forceinline__ v16bf load_a_f32(const float* A, int lda) {
  const int l = lane_id(), r = l & 15, h = l >> 4;
  v16bf a;
#pragma unroll
  for (int j = 0; j < 8; ++j) {
    const int bk = (j < 4) ? (h * 8 + 2 * j) : (16 + h * 8 + 2 * (j - 4));
    const float2 f = *(const float2*)(A + (size_t)r * lda + bk);
    a[2 * j]     = (__bf16)f.x;
    a[2 * j + 1] = (__bf16)f.y;
  }
  return a;
}

__device__ __forceinline__ v16bf load_a_bf16(const __bf16* A, int lda) {
  const int l = lane_id(), r = l & 15, h = l >> 4;
  v16bf a;
#pragma unroll
  for (int j = 0; j < 8; ++j) {
    const int bk = (j < 4) ? (h * 8 + 2 * j) : (16 + h * 8 + 2 * (j - 4));
    const __bf16* p = A + (size_t)r * lda + bk;
    a[2 * j]     = p[0];
    a[2 * j + 1] = p[1];
  }
  return a;
}

// B: 32x16 bf16 (KxN). lane l: col c = l&15, half h = l>>4.
// VGPR j (j<4): K = h*8 + 2j, +1 ; VGPR j (j>=4): K = 16 + h*8 + 2(j-4), +1.
__device__ __forceinline__ v16bf load_b_bf16(const __bf16* B, int sk, int sn) {
  const int l = lane_id(), c = l & 15, h = l >> 4;
  v16bf b;
#pragma unroll
  for (int j = 0; j < 8; ++j) {
    const int k = (j < 4) ? (h * 8 + 2 * j) : (16 + h * 8 + 2 * (j - 4));
    b[2 * j]     = B[(size_t)k * sk + (size_t)c * sn];
    b[2 * j + 1] = B[(size_t)(k + 1) * sk + (size_t)c * sn];
  }
  return b;
}

__device__ __forceinline__ v8f wmma_bf16(v16bf a, v16bf b, v8f c) {
  return __builtin_amdgcn_wmma_f32_16x16x32_bf16(false, a, false, b, (short)0, c,
                                                 false, false);
}

// Ordered float atomic max via int/uint atomics (works with -inf init).
__device__ __forceinline__ void atomicMaxF(float* addr, float val) {
  if (val >= 0.f) atomicMax((int*)addr, __float_as_int(val));
  else            atomicMin((unsigned int*)addr, (unsigned int)__float_as_int(val));
}

// ---------------------------------- kernels ---------------------------------

__global__ void init_kernel(float* __restrict__ mbuf, float* __restrict__ dbuf,
                            float* __restrict__ ubuf) {
  const long long i = (long long)blockIdx.x * blockDim.x + threadIdx.x;
  if (i < (long long)BATCHES * HDIM) ubuf[i] = 0.f;
  if (i < BATCHES) { mbuf[i] = -__builtin_inff(); dbuf[i] = 0.f; }
}

__global__ void f32_to_bf16_kernel(const float* __restrict__ src,
                                   __bf16* __restrict__ dst, int n) {
  const int i = blockIdx.x * blockDim.x + threadIdx.x;
  if (i < n) dst[i] = (__bf16)src[i];
}

// q = query @ Wq^T + bq  (M=B, N=128, K=128). One wave per 16x16 tile.
__global__ void qproj_kernel(const float* __restrict__ query,
                             const __bf16* __restrict__ Wq,
                             const float* __restrict__ bq,
                             float* __restrict__ qf, __bf16* __restrict__ qbf) {
  const int tm = blockIdx.x, tn = blockIdx.y;
  const float* A = query + (size_t)tm * 16 * QDIM;
  v8f acc = {};
#pragma unroll
  for (int k0 = 0; k0 < QDIM; k0 += 32) {
    v16bf a = load_a_f32(A + k0, QDIM);
    // B[k][n] = Wq[tn*16+n][k0+k]  -> sk=1, sn=QDIM (transposed weight)
    v16bf b = load_b_bf16(Wq + (size_t)(tn * 16) * QDIM + k0, 1, QDIM);
    acc = wmma_bf16(a, b, acc);
  }
  const int l = lane_id(), n = l & 15, h = l >> 4;
  const float bias = bq[tn * 16 + n];
#pragma unroll
  for (int i = 0; i < 8; ++i) {
    const int row = tm * 16 + i + 8 * h;
    const float v = acc[i] + bias;
    qf[(size_t)row * KDIM + tn * 16 + n]  = v;
    qbf[(size_t)row * KDIM + tn * 16 + n] = (__bf16)v;
  }
}

// p = (q @ Wk) * scale  (M=B, N=256, K=128)
__global__ void pproj_kernel(const __bf16* __restrict__ qbf,
                             const __bf16* __restrict__ Wk,
                             float* __restrict__ pbuf) {
  const int tm = blockIdx.x, tn = blockIdx.y;
  const __bf16* A = qbf + (size_t)tm * 16 * KDIM;
  v8f acc = {};
#pragma unroll
  for (int k0 = 0; k0 < KDIM; k0 += 32) {
    v16bf a = load_a_bf16(A + k0, KDIM);
    // B[k][n] = Wk[k0+k][tn*16+n] -> sk=HDIM, sn=1 (row-major)
    v16bf b = load_b_bf16(Wk + (size_t)k0 * HDIM + tn * 16, HDIM, 1);
    acc = wmma_bf16(a, b, acc);
  }
  const int l = lane_id(), n = l & 15, h = l >> 4;
#pragma unroll
  for (int i = 0; i < 8; ++i) {
    const int row = tm * 16 + i + 8 * h;
    pbuf[(size_t)row * HDIM + tn * 16 + n] = acc[i] * SCALE;
  }
}

// c[b] = (q[b] . bk) * scale
__global__ void cdot_kernel(const float* __restrict__ qf,
                            const float* __restrict__ bk,
                            float* __restrict__ cbuf) {
  const int w = threadIdx.x >> 5, l = threadIdx.x & 31;
  const int b = blockIdx.x * (blockDim.x >> 5) + w;
  if (b >= BATCHES) return;
  float acc = 0.f;
#pragma unroll
  for (int j = 0; j < 4; ++j) {
    const int k = l + 32 * j;
    acc = fmaf(qf[(size_t)b * KDIM + k], bk[k], acc);
  }
#pragma unroll
  for (int off = 16; off > 0; off >>= 1) acc += __shfl_xor(acc, off, 32);
  if (l == 0) cbuf[b] = acc * SCALE;
}

// s[n] = key[n] . p[batch[n]] + c[batch[n]]   (one wave per node; streams key)
__global__ void score_kernel(const float* __restrict__ key,
                             const float* __restrict__ pbuf,
                             const float* __restrict__ cbuf,
                             const int* __restrict__ batch,
                             float* __restrict__ sbuf, float* __restrict__ mbuf) {
  const int w = threadIdx.x >> 5, l = threadIdx.x & 31;
  const long long node = (long long)blockIdx.x * (blockDim.x >> 5) + w;
  if (node >= NNODES) return;
  const int b = batch[node];
  const float4* kp = (const float4*)(key  + (size_t)node * HDIM);
  const float4* pp = (const float4*)(pbuf + (size_t)b    * HDIM);
  const float4 k0 = kp[2 * l], k1 = kp[2 * l + 1];
  const float4 p0 = pp[2 * l], p1 = pp[2 * l + 1];
  float acc = k0.x * p0.x + k0.y * p0.y + k0.z * p0.z + k0.w * p0.w +
              k1.x * p1.x + k1.y * p1.y + k1.z * p1.z + k1.w * p1.w;
#pragma unroll
  for (int off = 16; off > 0; off >>= 1) acc += __shfl_xor(acc, off, 32);
  if (l == 0) {
    const float s = acc + cbuf[b];
    sbuf[node] = s;
    atomicMaxF(&mbuf[b], s);
  }
}

// e[n] = exp(s[n]-m[b]); denom[b] += e[n].  Sorted batch -> wave-segmented
// scan so only the last lane of each segment-run issues the atomic.
__global__ void exp_kernel(const float* __restrict__ sbuf,
                           const float* __restrict__ mbuf,
                           const int* __restrict__ batch,
                           float* __restrict__ ebuf, float* __restrict__ dbuf) {
  const long long n = (long long)blockIdx.x * blockDim.x + threadIdx.x;
  if (n >= NNODES) return;
  const int l = threadIdx.x & 31;
  const int b = batch[n];
  const float e = __expf(sbuf[n] - mbuf[b]);
  ebuf[n] = e;
  // inclusive segmented sum within wave (lanes are consecutive sorted nodes)
  float s = e;
#pragma unroll
  for (int off = 1; off < 32; off <<= 1) {
    const float o  = __shfl_up(s, off, 32);
    const int   ob = __shfl_up(b, off, 32);
    if (l >= off && ob == b) s += o;
  }
  const int nb = __shfl_down(b, 1, 32);
  if (l == 31 || nb != b) atomicAdd(&dbuf[b], s);
}

// u[b,:] += e[n] * value[n,:] — sorted-segment accumulation in registers.
// Block = 256 threads = 4 node-lane groups of 64 threads; thread owns 4
// contiguous dims (float4 / b128 loads). Group g walks nodes start+g,+4,...
// (still sorted), flushing its per-segment float4 on segment change.
#define NODES_PER_BLOCK 512
__global__ void uaccum_kernel(const float* __restrict__ value,
                              const float* __restrict__ ebuf,
                              const int* __restrict__ batch,
                              float* __restrict__ ubuf) {
  const int t = threadIdx.x;        // 0..255
  const int g = t >> 6;             // node-lane group 0..3 (wave-uniform)
  const int d = (t & 63) * 4;       // dim base 0..252
  const long long start = (long long)blockIdx.x * NODES_PER_BLOCK;
  const long long end   = start + NODES_PER_BLOCK;
  float4 acc = make_float4(0.f, 0.f, 0.f, 0.f);
  int curb = batch[start + g];
  for (long long n = start + g; n < end; n += 4) {
    if (n + 16 < NNODES)
      __builtin_prefetch(value + (size_t)(n + 16) * HDIM + d, 0, 1);
    const int b = batch[n];
    if (b != curb) {
      float* u = ubuf + (size_t)curb * HDIM + d;
      atomicAdd(u + 0, acc.x); atomicAdd(u + 1, acc.y);
      atomicAdd(u + 2, acc.z); atomicAdd(u + 3, acc.w);
      acc = make_float4(0.f, 0.f, 0.f, 0.f);
      curb = b;
    }
    const float  e = ebuf[n];
    const float4 v = *(const float4*)(value + (size_t)n * HDIM + d);
    acc.x = fmaf(e, v.x, acc.x);
    acc.y = fmaf(e, v.y, acc.y);
    acc.z = fmaf(e, v.z, acc.z);
    acc.w = fmaf(e, v.w, acc.w);
  }
  float* u = ubuf + (size_t)curb * HDIM + d;
  atomicAdd(u + 0, acc.x); atomicAdd(u + 1, acc.y);
  atomicAdd(u + 2, acc.z); atomicAdd(u + 3, acc.w);
}

// num = u @ Wv^T + denom*bv ; a = num/denom (guarded), stored bf16
__global__ void nproj_kernel(const float* __restrict__ ubuf,
                             const __bf16* __restrict__ Wv,
                             const float* __restrict__ bv,
                             const float* __restrict__ dbuf,
                             __bf16* __restrict__ abf) {
  const int tm = blockIdx.x, tn = blockIdx.y;
  const float* A = ubuf + (size_t)tm * 16 * HDIM;
  v8f acc = {};
#pragma unroll
  for (int k0 = 0; k0 < HDIM; k0 += 32) {
    v16bf a = load_a_f32(A + k0, HDIM);
    v16bf b = load_b_bf16(Wv + (size_t)(tn * 16) * HDIM + k0, 1, HDIM);
    acc = wmma_bf16(a, b, acc);
  }
  const int l = lane_id(), n = l & 15, h = l >> 4;
  const float bvn = bv[tn * 16 + n];
#pragma unroll
  for (int i = 0; i < 8; ++i) {
    const int row = tm * 16 + i + 8 * h;
    const float den = dbuf[row];
    const float num = acc[i] + den * bvn;
    const float a = (den > 0.f) ? num / fmaxf(den, 1e-30f) : 0.f;
    abf[(size_t)row * VDIM + tn * 16 + n] = (__bf16)a;
  }
}

// out = a @ Wo^T + bo
__global__ void out_kernel(const __bf16* __restrict__ abf,
                           const __bf16* __restrict__ Wo,
                           const float* __restrict__ bo,
                           float* __restrict__ out) {
  const int tm = blockIdx.x, tn = blockIdx.y;
  const __bf16* A = abf + (size_t)tm * 16 * VDIM;
  v8f acc = {};
#pragma unroll
  for (int k0 = 0; k0 < VDIM; k0 += 32) {
    v16bf a = load_a_bf16(A + k0, VDIM);
    v16bf b = load_b_bf16(Wo + (size_t)(tn * 16) * VDIM + k0, 1, VDIM);
    acc = wmma_bf16(a, b, acc);
  }
  const int l = lane_id(), n = l & 15, h = l >> 4;
  const float bias = bo[tn * 16 + n];
#pragma unroll
  for (int i = 0; i < 8; ++i) {
    const int row = tm * 16 + i + 8 * h;
    out[(size_t)row * VDIM + tn * 16 + n] = acc[i] + bias;
  }
}

// --------------------------------- launcher ---------------------------------

extern "C" void kernel_launch(void* const* d_in, const int* in_sizes, int n_in,
                              void* d_out, int out_size, void* d_ws, size_t ws_size,
                              hipStream_t stream) {
  const float* query = (const float*)d_in[0];
  const float* key   = (const float*)d_in[1];
  const float* value = (const float*)d_in[2];
  const int*   batch = (const int*)d_in[3];
  const float* Wq    = (const float*)d_in[4];
  const float* bq    = (const float*)d_in[5];
  const float* Wk    = (const float*)d_in[6];
  const float* bk    = (const float*)d_in[7];
  const float* Wv    = (const float*)d_in[8];
  const float* bv    = (const float*)d_in[9];
  const float* Wo    = (const float*)d_in[10];
  const float* bo    = (const float*)d_in[11];
  float* out = (float*)d_out;

  // ---- workspace carve (floats) ----
  float* w = (float*)d_ws;
  size_t o = 0;
  float* qf   = w + o; o += (size_t)BATCHES * KDIM;   // q (f32)
  float* pbuf = w + o; o += (size_t)BATCHES * HDIM;   // p = q@Wk * scale
  float* cbuf = w + o; o += BATCHES;                  // c = q.bk * scale
  float* mbuf = w + o; o += BATCHES;                  // segment max
  float* dbuf = w + o; o += BATCHES;                  // denom
  float* sbuf = w + o; o += NNODES;                   // scores
  float* ebuf = w + o; o += NNODES;                   // exp weights
  float* ubuf = w + o; o += (size_t)BATCHES * HDIM;   // u = sum e*value
  __bf16* bfb = (__bf16*)(w + o);
  size_t ob = 0;
  __bf16* qbf = bfb + ob; ob += (size_t)BATCHES * KDIM;
  __bf16* abf = bfb + ob; ob += (size_t)BATCHES * VDIM;
  __bf16* WqB = bfb + ob; ob += KDIM * QDIM;
  __bf16* WkB = bfb + ob; ob += KDIM * HDIM;
  __bf16* WvB = bfb + ob; ob += VDIM * HDIM;
  __bf16* WoB = bfb + ob; ob += VDIM * VDIM;

  // 1. init accumulators
  init_kernel<<<(BATCHES * HDIM + 255) / 256, 256, 0, stream>>>(mbuf, dbuf, ubuf);

  // 2. weights -> bf16
  f32_to_bf16_kernel<<<(KDIM * QDIM + 255) / 256, 256, 0, stream>>>(Wq, WqB, KDIM * QDIM);
  f32_to_bf16_kernel<<<(KDIM * HDIM + 255) / 256, 256, 0, stream>>>(Wk, WkB, KDIM * HDIM);
  f32_to_bf16_kernel<<<(VDIM * HDIM + 255) / 256, 256, 0, stream>>>(Wv, WvB, VDIM * HDIM);
  f32_to_bf16_kernel<<<(VDIM * VDIM + 255) / 256, 256, 0, stream>>>(Wo, WoB, VDIM * VDIM);

  // 3. q projection (WMMA)
  qproj_kernel<<<dim3(BATCHES / 16, KDIM / 16), 32, 0, stream>>>(query, WqB, bq, qf, qbf);

  // 4. c = q.bk * scale ; p = q@Wk * scale (WMMA)
  cdot_kernel<<<BATCHES / 8, 256, 0, stream>>>(qf, bk, cbuf);
  pproj_kernel<<<dim3(BATCHES / 16, HDIM / 16), 32, 0, stream>>>(qbf, WkB, pbuf);

  // 5. scores + segment max (streams key, 536 MB)
  score_kernel<<<NNODES / 8, 256, 0, stream>>>(key, pbuf, cbuf, batch, sbuf, mbuf);

  // 6. e = exp(s-m), denom (wave-segmented atomics)
  exp_kernel<<<NNODES / 256, 256, 0, stream>>>(sbuf, mbuf, batch, ebuf, dbuf);

  // 7. u accumulation (streams value, 536 MB, b128 per lane)
  uaccum_kernel<<<NNODES / NODES_PER_BLOCK, 256, 0, stream>>>(value, ebuf, batch, ubuf);

  // 8. num = u@Wv^T + denom*bv ; a = num/denom (WMMA)
  nproj_kernel<<<dim3(BATCHES / 16, VDIM / 16), 32, 0, stream>>>(ubuf, WvB, bv, dbuf, abf);

  // 9. out = a@Wo^T + bo (WMMA)
  out_kernel<<<dim3(BATCHES / 16, VDIM / 16), 32, 0, stream>>>(abf, WoB, bo, out);
}
